// RPEMultiHeadAttention_70987219468990
// MI455X (gfx1250) — matrix operand order repaired and checked by reading, not compiled, not run on hardware
//
#include <hip/hip_runtime.h>
#include <hip/hip_bf16.h>

#define NN 512
#define MM 512
#define CC 256
#define HH 4
#define DHD 64

// padded LDS row strides (in bf16 elements): +8 bf16 = +16B keeps b128 alignment
// and shifts consecutive rows by 4 LDS banks (stride 528B = 132 banks mod 64 = 4).
#define CCP 264   // E rows (512 x 264) and qpb rows (16 x 264)
#define ATP 520   // atb rows (16 x 520)

typedef __bf16 bf16_t;
typedef __attribute__((ext_vector_type(16))) __bf16 v16bf;
typedef __attribute__((ext_vector_type(8)))  __bf16 v8bf;
typedef __attribute__((ext_vector_type(4)))  __bf16 v4bf;
typedef __attribute__((ext_vector_type(8)))  float   v8f;

// ---------------- Prologue kernels (tiny vs. main pass; VALU is fine) -------

// q/k/v projections: out[n,c] = in[n,:]·W[c,:] + b[c]
__global__ void proj_kernel(const float* __restrict__ inq, const float* __restrict__ ink,
                            const float* __restrict__ inv,
                            const float* __restrict__ Wq, const float* __restrict__ bq,
                            const float* __restrict__ Wk, const float* __restrict__ bk,
                            const float* __restrict__ Wv, const float* __restrict__ bv,
                            float* __restrict__ q, float* __restrict__ k, float* __restrict__ v) {
  __shared__ float row[CC];
  const int n = blockIdx.x;
  const int c = threadIdx.x;
  const int which = blockIdx.y;
  const float* in = which == 0 ? inq : which == 1 ? ink : inv;
  const float* W  = which == 0 ? Wq  : which == 1 ? Wk  : Wv;
  const float* b  = which == 0 ? bq  : which == 1 ? bk  : bv;
  float* out      = which == 0 ? q   : which == 1 ? k   : v;
  row[c] = in[n * CC + c];
  __syncthreads();
  const float4* W4 = (const float4*)(W + c * CC);
  const float4* r4 = (const float4*)row;
  float acc = 0.f;
#pragma unroll 8
  for (int i = 0; i < CC / 4; ++i) {
    float4 w = W4[i], x = r4[i];
    acc += w.x * x.x + w.y * x.y + w.z * x.z + w.w * x.w;
  }
  out[n * CC + c] = acc + b[c];
}

// q~[n,h,i] = sum_d q[n, h*64+d] * Wp[h*64+d, i]   (fold Wp into q)
__global__ void qp_kernel(const float* __restrict__ q, const float* __restrict__ Wp,
                          float* __restrict__ qp) {
  __shared__ float qrow[DHD];
  const int bid = blockIdx.x;           // n*4 + h
  const int n = bid >> 2, h = bid & 3;
  const int i = threadIdx.x;            // 0..255
  if (i < DHD) qrow[i] = q[n * CC + h * DHD + i];
  __syncthreads();
  float acc = 0.f;
#pragma unroll 8
  for (int d = 0; d < DHD; ++d) acc += qrow[d] * Wp[(h * DHD + d) * CC + i];
  qp[(n * HH + h) * CC + i] = acc;
}

// qbp[n,h] = q[h,n,:]·bp_h  (constant additive term from the bp bias)
__global__ void qbp_kernel(const float* __restrict__ q, const float* __restrict__ bp,
                           float* __restrict__ qbp) {
  const int idx = blockIdx.x * blockDim.x + threadIdx.x;  // 0..2047
  if (idx >= NN * HH) return;
  const int n = idx >> 2, h = idx & 3;
  float acc = 0.f;
  for (int d = 0; d < DHD; ++d) acc += q[n * CC + h * DHD + d] * bp[h * DHD + d];
  qbp[idx] = acc;
}

// scores_e[n,h,m] = q_h[n,:]·k_h[m,:]
__global__ void se_kernel(const float* __restrict__ q, const float* __restrict__ k,
                          float* __restrict__ se) {
  __shared__ float qrow[DHD];
  const int n = blockIdx.x, h = blockIdx.y;
  const int m = threadIdx.x;            // 0..511
  if (m < DHD) qrow[m] = q[n * CC + h * DHD + m];
  __syncthreads();
  const float4* k4 = (const float4*)(k + m * CC + h * DHD);
  const float4* q4 = (const float4*)qrow;
  float acc = 0.f;
#pragma unroll
  for (int i = 0; i < DHD / 4; ++i) {
    float4 a = k4[i], b = q4[i];
    acc += a.x * b.x + a.y * b.y + a.z * b.z + a.w * b.w;
  }
  se[((size_t)n * HH + h) * MM + m] = acc;
}

// ---------------- Fused main kernel: one block (16 waves) per query row n ---
// LDS (dynamic), all row-padded to kill bank conflicts:
//   E   : bf16[512*264] = 270336 B   (embed_qk row-tile, read from HBM once)
//   sc  : f32 [4*512]   =   8192 B   (raw scores -> normalized attn)
//   qpb : bf16[16*264]  =   8448 B   (q~ as WMMA B operand, rows 4..15 zero)
//   atb : bf16[16*520]  =  16640 B   (attn as WMMA A operand, rows 4..15 zero)
//   U   : f32 [4*256]   =   4096 B   (attn @ E)
//   red : f32 [512]     =   2048 B   (reductions)
#define LDS_E    0
#define LDS_SC   (270336)
#define LDS_QPB  (270336 + 8192)
#define LDS_ATB  (270336 + 8192 + 8448)
#define LDS_U    (270336 + 8192 + 8448 + 16640)
#define LDS_RED  (270336 + 8192 + 8448 + 16640 + 4096)
#define LDS_TOT  (270336 + 8192 + 8448 + 16640 + 4096 + 2048)   // 309760 <= 320KB

__global__ __launch_bounds__(512)
void rpe_attn_main(const float* __restrict__ embed,
                   const float* __restrict__ qp, const float* __restrict__ se,
                   const float* __restrict__ qbp,
                   const float* __restrict__ vw,
                   const float* __restrict__ Wvp, const float* __restrict__ bvp,
                   float* __restrict__ out_hidden, float* __restrict__ out_attn) {
  extern __shared__ char smem[];
  bf16_t* E   = (bf16_t*)(smem + LDS_E);
  float*  sc  = (float*) (smem + LDS_SC);
  bf16_t* qpb = (bf16_t*)(smem + LDS_QPB);
  bf16_t* atb = (bf16_t*)(smem + LDS_ATB);
  float*  U   = (float*) (smem + LDS_U);
  float*  red = (float*) (smem + LDS_RED);

  const int n    = blockIdx.x;
  const int t    = threadIdx.x;        // 0..511
  const int lane = t & 31;
  const int wave = t >> 5;             // 0..15

  // ---- Phase 1: stream E_n (512KB f32) -> bf16 LDS; deep load pipeline -----
  {
    const float4* src = (const float4*)(embed + (size_t)n * MM * CC);
    for (int j0 = 0; j0 < 64; j0 += 8) {           // 64 float4 per thread
      float4 f[8];
#pragma unroll
      for (int u = 0; u < 8; ++u) f[u] = src[(j0 + u) * 512 + t];
#pragma unroll
      for (int u = 0; u < 8; ++u) {
        const int idx = (j0 + u) * 512 + t;        // float4 index (64 per row)
        const int m = idx >> 6;
        const int i = (idx & 63) * 4;
        v4bf b4;
        b4[0] = (bf16_t)f[u].x; b4[1] = (bf16_t)f[u].y;
        b4[2] = (bf16_t)f[u].z; b4[3] = (bf16_t)f[u].w;
        *(v4bf*)(E + m * CCP + i) = b4;
      }
    }
  }
  {
    const float* qpn = qp + (size_t)n * HH * CC;
    for (int ii = t; ii < 16 * CCP; ii += 512) {
      const int h = ii / CCP, i = ii % CCP;
      const float val = (h < HH && i < CC) ? qpn[h * CC + i] : 0.f;
      qpb[ii] = (bf16_t)val;
    }
    for (int ii = t; ii < 16 * ATP; ii += 512) atb[ii] = (bf16_t)0.f;
  }
  __syncthreads();

  // ---- Phase 2: scores = (E_n @ q~^T + se + q·bp) / 8  via WMMA ------------
  const float* sen = se + (size_t)n * HH * MM;
  for (int mt = wave * 2; mt < wave * 2 + 2; ++mt) {
    const int m0 = mt * 16;
    v8f acc = {};
    for (int kk = 0; kk < CC / 32; ++kk) {
      const int k0 = kk * 32;
      // A fragment: 16x32 bf16, rows of E
      const int arow  = m0 + (lane & 15);
      const int abase = k0 + (lane >> 4) * 8;
      v8bf alo = *(const v8bf*)(E + arow * CCP + abase);
      v8bf ahi = *(const v8bf*)(E + arow * CCP + abase + 16);
      v16bf A;
#pragma unroll
      for (int x = 0; x < 8; ++x) { A[x] = alo[x]; A[x + 8] = ahi[x]; }
      // B fragment: 32x16 bf16, columns = heads (q~)
      const int bcol  = lane & 15;
      const int bbase = k0 + (lane >> 4) * 8;
      v8bf blo = *(const v8bf*)(qpb + bcol * CCP + bbase);
      v8bf bhi = *(const v8bf*)(qpb + bcol * CCP + bbase + 16);
      v16bf B;
#pragma unroll
      for (int x = 0; x < 8; ++x) { B[x] = blo[x]; B[x + 8] = bhi[x]; }
      acc = __builtin_amdgcn_wmma_f32_16x16x32_bf16(false, A, false, B, (short)0, acc,
                                                    false, false);
    }
    const int head = lane & 15;
    if (head < HH) {
      const float qb = qbp[n * HH + head];
#pragma unroll
      for (int r = 0; r < 8; ++r) {
        const int m = m0 + r + (lane >> 4) * 8;
        sc[head * MM + m] = (acc[r] + sen[head * MM + m] + qb) * 0.125f;
      }
    }
  }
  __syncthreads();

  // ---- Phase 3: softmax per head over m; write attn (f32 out + bf16 frag) --
  {
    const int h = t >> 7, g = t & 127;
    float mx = -3.4e38f;
#pragma unroll
    for (int j = 0; j < 4; ++j) mx = fmaxf(mx, sc[h * MM + g + j * 128]);
    red[t] = mx; __syncthreads();
    for (int s = 64; s > 0; s >>= 1) {
      if (g < s) red[t] = fmaxf(red[t], red[t + s]);
      __syncthreads();
    }
    const float hmax = red[h << 7];
    __syncthreads();
    float sm = 0.f;
#pragma unroll
    for (int j = 0; j < 4; ++j) {
      const int m = g + j * 128;
      const float e = __expf(sc[h * MM + m] - hmax);
      sc[h * MM + m] = e;
      sm += e;
    }
    red[t] = sm; __syncthreads();
    for (int s = 64; s > 0; s >>= 1) {
      if (g < s) red[t] += red[t + s];
      __syncthreads();
    }
    const float inv = 1.f / red[h << 7];
#pragma unroll
    for (int j = 0; j < 4; ++j) {
      const int m = g + j * 128;
      const float a = sc[h * MM + m] * inv;
      sc[h * MM + m]  = a;                 // keep f32 attn for h0
      atb[h * ATP + m] = (bf16_t)a;        // WMMA A operand for U
      out_attn[((size_t)h * NN + n) * MM + m] = a;
    }
  }
  __syncthreads();

  // ---- Phase 4: U = attn @ E_n  (4x512 @ 512x256) via WMMA -----------------
  {
    const int nt = wave;                  // one 16-col tile per wave
    const int col0 = nt * 16;
    v8f acc = {};
    for (int kk = 0; kk < MM / 32; ++kk) {
      const int k0 = kk * 32;
      const int arow  = lane & 15;
      const int abase = k0 + (lane >> 4) * 8;
      v8bf alo = *(const v8bf*)(atb + arow * ATP + abase);
      v8bf ahi = *(const v8bf*)(atb + arow * ATP + abase + 16);
      v16bf A;
#pragma unroll
      for (int x = 0; x < 8; ++x) { A[x] = alo[x]; A[x + 8] = ahi[x]; }
      // B fragment: 32x16 slice of E (rows = m, cols = channel i); padded gather
      const int bcol = col0 + (lane & 15);
      const int bk0  = k0 + (lane >> 4) * 8;
      v16bf B;
#pragma unroll
      for (int x = 0; x < 16; ++x) B[x] = E[(bk0 + (x & 7) + (x >> 3) * 16) * CCP + bcol];
      acc = __builtin_amdgcn_wmma_f32_16x16x32_bf16(false, A, false, B, (short)0, acc,
                                                    false, false);
    }
    const int colc = col0 + (lane & 15);
#pragma unroll
    for (int r = 0; r < 8; ++r) {
      const int row = r + (lane >> 4) * 8;
      if (row < HH) U[row * CC + colc] = acc[r];
    }
  }
  __syncthreads();

  // ---- Phase 5: hidden = attn@v + Wvp_h @ U + bvp --------------------------
  if (t < 256) {
    const int h = t >> 6;        // c == t == h*64 + d
    float h0 = 0.f;
    for (int m = 0; m < MM; ++m) h0 += sc[h * MM + m] * vw[m * CC + t];
    float h1 = bvp[t];
    const float4* w4 = (const float4*)(Wvp + t * CC);
    const float4* u4 = (const float4*)(U + h * CC);
#pragma unroll 8
    for (int i = 0; i < CC / 4; ++i) {
      float4 w = w4[i], u = u4[i];
      h1 += w.x * u.x + w.y * u.y + w.z * u.z + w.w * u.w;
    }
    out_hidden[n * CC + t] = h0 + h1;
  }
}

// ---------------------------------------------------------------------------
extern "C" void kernel_launch(void* const* d_in, const int* in_sizes, int n_in,
                              void* d_out, int out_size, void* d_ws, size_t ws_size,
                              hipStream_t stream) {
  const float* input_q = (const float*)d_in[0];
  const float* input_k = (const float*)d_in[1];
  const float* input_v = (const float*)d_in[2];
  const float* embed   = (const float*)d_in[3];
  const float* Wq = (const float*)d_in[4];   const float* bq  = (const float*)d_in[5];
  const float* Wk = (const float*)d_in[6];   const float* bk  = (const float*)d_in[7];
  const float* Wv = (const float*)d_in[8];   const float* bv  = (const float*)d_in[9];
  const float* Wp = (const float*)d_in[10];  const float* bp  = (const float*)d_in[11];
  const float* Wvp = (const float*)d_in[12]; const float* bvp = (const float*)d_in[13];

  float* ws     = (float*)d_ws;
  float* q_ws   = ws;                               // 512*256
  float* k_ws   = q_ws  + NN * CC;                  // 512*256
  float* v_ws   = k_ws  + MM * CC;                  // 512*256
  float* qp_ws  = v_ws  + MM * CC;                  // 512*4*256
  float* se_ws  = qp_ws + NN * HH * CC;             // 512*4*512
  float* qbp_ws = se_ws + (size_t)NN * HH * MM;     // 512*4

  float* out_hidden = (float*)d_out;                // (1,512,256)
  float* out_attn   = out_hidden + NN * CC;         // (1,4,512,512)

  proj_kernel<<<dim3(NN, 3), 256, 0, stream>>>(input_q, input_k, input_v,
                                               Wq, bq, Wk, bk, Wv, bv,
                                               q_ws, k_ws, v_ws);
  qp_kernel<<<NN * HH, 256, 0, stream>>>(q_ws, Wp, qp_ws);
  qbp_kernel<<<(NN * HH + 255) / 256, 256, 0, stream>>>(q_ws, bp, qbp_ws);
  se_kernel<<<dim3(NN, HH), 512, 0, stream>>>(q_ws, k_ws, se_ws);

  (void)hipFuncSetAttribute(reinterpret_cast<const void*>(rpe_attn_main),
                            hipFuncAttributeMaxDynamicSharedMemorySize, LDS_TOT);
  rpe_attn_main<<<NN, 512, LDS_TOT, stream>>>(embed, qp_ws, se_ws, qbp_ws, v_ws,
                                              Wvp, bvp, out_hidden, out_attn);
}
